// LGNLayer_51007031607532
// MI455X (gfx1250) — matrix operand order repaired and checked by reading, not compiled, not run on hardware
//
#include <hip/hip_runtime.h>
#include <hip/hip_bf16.h>

#define NN 8192
#define TT 64

typedef __attribute__((ext_vector_type(16))) __bf16 v16bf;
typedef __attribute__((ext_vector_type(8)))  __bf16 v8bf;
typedef __attribute__((ext_vector_type(4)))  __bf16 v4bf;
typedef __attribute__((ext_vector_type(8)))  float  v8f;

// ---------------------------------------------------------------------------
// Init: convert W (f32, 256MB) -> bf16 (128MB, fits 192MB L2), copy v state,
// convert firing_init to bf16 into double-buffer slot 0.
// Grid is exactly NN*NN/4 threads, 4 elements per thread.
// ---------------------------------------------------------------------------
__global__ __launch_bounds__(256) void lgn_init(const float* __restrict__ W,
                                                const float* __restrict__ v0,
                                                const float* __restrict__ f0,
                                                __bf16* __restrict__ Wb,
                                                float* __restrict__ v,
                                                __bf16* __restrict__ fb0) {
    size_t i = (size_t)blockIdx.x * blockDim.x + threadIdx.x;
    float4 w = ((const float4*)W)[i];
    v4bf o;
    o[0] = (__bf16)w.x; o[1] = (__bf16)w.y;
    o[2] = (__bf16)w.z; o[3] = (__bf16)w.w;
    ((v4bf*)Wb)[i] = o;
    if (i < NN) {
        v[i]   = v0[i];
        fb0[i] = (__bf16)f0[i];
    }
}

// ---------------------------------------------------------------------------
// One LIF time step:
//   node_x = W(bf16) @ firing(bf16)  via v_wmma_f32_16x16x32_bf16
//   v = v*0.99 + node_x + ext[t]; fire = (v>=1); v *= (1-fire)
// Each wave owns 16 rows; block = 8 waves = 128 rows; grid = 64 blocks.
// B operand = firing vector staged in LDS, replicated across all 16 columns.
// ---------------------------------------------------------------------------
__global__ __launch_bounds__(256) void lgn_step(const __bf16* __restrict__ Wb,
                                                const float* __restrict__ ext,
                                                float* __restrict__ v,
                                                const __bf16* __restrict__ fin,
                                                __bf16* __restrict__ fout,
                                                float* __restrict__ out,
                                                int t) {
    __shared__ __bf16 sF[NN];          // 16 KB: previous-step firing vector

    const int tid = threadIdx.x;

    // Cooperative 128-bit staging of firing vector into LDS.
    {
        const v8bf* g = (const v8bf*)fin;
        v8bf*       s = (v8bf*)sF;
        #pragma unroll
        for (int i = 0; i < (NN / 8) / 256; ++i)
            s[tid + i * 256] = g[tid + i * 256];
    }
    __syncthreads();

    const int lane    = tid & 31;
    const int wave    = tid >> 5;
    const int rowBase = blockIdx.x * 128 + wave * 16;

    // A-operand addressing, contiguous k order (same as B below):
    //   lanes 0-15  : row = rowBase+lane,      K chunks {0..7},{8..15}
    //   lanes 16-31 : row = rowBase+lane-16,   K chunks {16..23},{24..31}
    const int arow = rowBase + (lane & 15);
    const __bf16* __restrict__ ar = Wb + (size_t)arow * NN;
    const int hi  = lane & 16;          // 0 or 16
    const int kA0 = hi ? 16 : 0;
    const int kA1 = hi ? 24 : 8;

    v8f c = {};
    for (int k0 = 0; k0 < NN; k0 += 32) {
        __builtin_prefetch(ar + k0 + 512, 0, 0);     // global_prefetch_b8 ahead on W stream
        v8bf a0 = *(const v8bf*)(ar + k0 + kA0);     // 16B global load
        v8bf a1 = *(const v8bf*)(ar + k0 + kA1);     // 16B global load
        v16bf a = __builtin_shufflevector(a0, a1,
                    0,1,2,3,4,5,6,7,8,9,10,11,12,13,14,15);
        // B: lanes 0-15 hold K=k0..k0+15, lanes 16-31 hold K=k0+16..k0+31
        // (all 16 columns identical = firing vector)
        v16bf b = *(const v16bf*)(sF + k0 + hi);     // 32B LDS load
        c = __builtin_amdgcn_wmma_f32_16x16x32_bf16(
                /*neg_a=*/false, a, /*neg_b=*/false, b,
                /*c_mod=*/(short)0, c, /*reuse_a=*/false, /*reuse_b=*/false);
    }

    // D layout: lane 0 holds D[m][0] for m=0..7, lane 16 holds m=8..15.
    if ((lane & 15) == 0) {
        const int mbase = rowBase + (hi ? 8 : 0);
        #pragma unroll
        for (int j = 0; j < 8; ++j) {
            const int r = mbase + j;
            float vv   = v[r] * 0.99f + c[j] + ext[(size_t)t * NN + r];
            float fire = (vv >= 1.0f) ? 1.0f : 0.0f;
            vv *= (1.0f - fire);
            v[r] = vv;
            out[(size_t)t * NN + r] = fire;
            fout[r] = (__bf16)fire;     // next step's B operand
        }
    }
}

// ---------------------------------------------------------------------------
// Inputs: d_in[0]=W [N*N f32], d_in[1]=v_init [N f32], d_in[2]=firing_init
// [N f32], d_in[3]=ext [T*N f32]. Output: firing_matrix [T*N f32].
// Workspace: Wb bf16 (128MB) | v f32 (32KB) | firing double-buffer bf16 (32KB)
// ---------------------------------------------------------------------------
extern "C" void kernel_launch(void* const* d_in, const int* in_sizes, int n_in,
                              void* d_out, int out_size, void* d_ws, size_t ws_size,
                              hipStream_t stream) {
    const float* W   = (const float*)d_in[0];
    const float* v0  = (const float*)d_in[1];
    const float* f0  = (const float*)d_in[2];
    const float* ext = (const float*)d_in[3];
    float*       out = (float*)d_out;

    char*   ws = (char*)d_ws;
    __bf16* Wb = (__bf16*)ws;                                    // 128 MB
    float*  v  = (float*)(ws + (size_t)NN * NN * 2);             // 32 KB
    __bf16* fb = (__bf16*)(ws + (size_t)NN * NN * 2 + NN * 4);   // 2 x 16 KB

    // 1) one-time (per launch) convert + state init
    lgn_init<<<(NN / 4) * (NN / 256), 256, 0, stream>>>(W, v0, f0, Wb, v, fb);

    // 2) 64 sequential LIF steps; stream ordering carries the recurrence.
    for (int t = 0; t < TT; ++t) {
        const __bf16* fin  = fb + (t & 1) * NN;        // firing from step t-1
        __bf16*       fout = fb + ((t + 1) & 1) * NN;  // firing of step t
        lgn_step<<<NN / 128, 256, 0, stream>>>(Wb, ext, v, fin, fout, out, t);
    }
}
